// SelfAttentionModel_54004918780603
// MI455X (gfx1250) — matrix operand order repaired and checked by reading, not compiled, not run on hardware
//
#include <hip/hip_runtime.h>
#include <stdint.h>

#define B_  4
#define N_  4096      // H*W
#define C_  256
#define CQ_ 32
#define AT_WAVES 4
#define KEYS 64
#define KITERS (N_ / KEYS)

typedef __bf16 bf16_t;
typedef __attribute__((ext_vector_type(16))) __bf16 v16bf;
typedef __attribute__((ext_vector_type(8)))  float  v8f;
typedef __attribute__((ext_vector_type(2)))  __bf16 v2bf;

union FragBF { v16bf v; uint4 q[2]; uint32_t u[8]; };

// Native bf16 conversions (gfx1250 has hardware bf16 cvt ops).
__device__ __forceinline__ uint32_t pk2u(float a, float b) {
    v2bf t;
    t[0] = (bf16_t)a;
    t[1] = (bf16_t)b;
    return __builtin_bit_cast(uint32_t, t);
}
__device__ __forceinline__ uint16_t bf1(float a) {
    bf16_t h = (bf16_t)a;
    return __builtin_bit_cast(uint16_t, h);
}

// async global->LDS, 16B per lane, tracked by ASYNCcnt
#define AB128(LDS, G, OFFSTR)                                               \
    asm volatile("global_load_async_to_lds_b128 %0, %1, off offset:" OFFSTR \
                 :: "v"(LDS), "v"(G) : "memory")

__device__ __forceinline__ void stage_tile(const unsigned short* ktg, unsigned short* ktl,
                                           const unsigned short* vg,  unsigned short* vl)
{
    uint32_t kl = (uint32_t)(uint64_t)ktl;   // low 32 bits of generic = LDS offset
    uint32_t vo = (uint32_t)(uint64_t)vl;
    AB128(kl, ktg, "0");   AB128(kl, ktg, "16");       // 32B of the KT tile row
    AB128(vo, vg, "0");    AB128(vo, vg, "16");        // 256B of the V tile row
    AB128(vo, vg, "32");   AB128(vo, vg, "48");
    AB128(vo, vg, "64");   AB128(vo, vg, "80");
    AB128(vo, vg, "96");   AB128(vo, vg, "112");
    AB128(vo, vg, "128");  AB128(vo, vg, "144");
    AB128(vo, vg, "160");  AB128(vo, vg, "176");
    AB128(vo, vg, "192");  AB128(vo, vg, "208");
    AB128(vo, vg, "224");  AB128(vo, vg, "240");
}

// ---------------------------------------------------------------------------
// One-pass fp32 -> bf16 conversion (8 elems / thread).
// ---------------------------------------------------------------------------
__global__ __launch_bounds__(256)
void cvt_bf16_kernel(const float* __restrict__ src,
                     unsigned short* __restrict__ dst, int n8)
{
    const int i = blockIdx.x * blockDim.x + threadIdx.x;
    if (i >= n8) return;
    const float4 f0 = ((const float4*)src)[2 * i];
    const float4 f1 = ((const float4*)src)[2 * i + 1];
    uint4 o;
    o.x = pk2u(f0.x, f0.y);
    o.y = pk2u(f0.z, f0.w);
    o.z = pk2u(f1.x, f1.y);
    o.w = pk2u(f1.z, f1.w);
    ((uint4*)dst)[i] = o;
}

// ---------------------------------------------------------------------------
// QKV projection (all-bf16 operands): out = bf16( Xbf @ Wbf + bias )
// mode 0: row-major. mode 1: transposed KT[B][CQ][N].
// Pure load->WMMA inner loop, no conversions.
// ---------------------------------------------------------------------------
__global__ __launch_bounds__(256)
void qkv_proj_kernel(const unsigned short* __restrict__ xbf,
                     const unsigned short* __restrict__ wbf,
                     const float* __restrict__ bias,
                     unsigned short* __restrict__ out,
                     int ncols, int mode)
{
    const int lane = threadIdx.x & 31;
    const int wave = threadIdx.x >> 5;
    const int m0   = (blockIdx.x * (blockDim.x >> 5) + wave) * 16;
    const int c0   = blockIdx.y * 16;
    const int half = lane >> 4;
    const int l15  = lane & 15;

    v8f acc = (v8f){0.f,0.f,0.f,0.f,0.f,0.f,0.f,0.f};

    const unsigned short* xrow = xbf + (size_t)(m0 + l15) * C_;
    for (int k0 = 0; k0 < C_; k0 += 32) {
        FragBF a, b;
        const int kA = k0 + half * 8;                 // A swizzle chunks: kA, kA+16
        a.q[0] = *(const uint4*)(xrow + kA);
        a.q[1] = *(const uint4*)(xrow + kA + 16);
        for (int s = 0; s < 16; ++s)
            b.v[s] = __builtin_bit_cast(bf16_t, wbf[(size_t)(k0 + half * 8 + (s & 7) + ((s >> 3) << 4)) * ncols + c0 + l15]);
        acc = __builtin_amdgcn_wmma_f32_16x16x32_bf16(
                  false, a.v, false, b.v, (short)0, acc, false, false);
    }

    const float bsc = bias[c0 + l15];
    if (mode == 0) {
#pragma unroll
        for (int r = 0; r < 8; ++r) {
            const int m = m0 + r + 8 * half;
            out[(size_t)m * ncols + c0 + l15] = bf1(acc[r] + bsc);
        }
    } else {                                          // KT[B][CQ][N]
#pragma unroll
        for (int r = 0; r < 8; ++r) {
            const int tok = m0 + r + 8 * half;
            const int bb  = tok >> 12;
            const int n   = tok & (N_ - 1);
            out[((size_t)bb * CQ_ + (c0 + l15)) * N_ + n] = bf1(acc[r] + bsc);
        }
    }
}

// ---------------------------------------------------------------------------
// Flash attention, 4 waves/block, 16 q/wave, 64-key chunks, double-buffered
// async-to-LDS K/V staging.
// ---------------------------------------------------------------------------
__global__ __launch_bounds__(128)
void attn_kernel(const unsigned short* __restrict__ Q,
                 const unsigned short* __restrict__ KT,
                 const unsigned short* __restrict__ V,
                 const float* __restrict__ inputs,
                 const float* __restrict__ gammaPtr,
                 float* __restrict__ out)
{
    __shared__ alignas(16) unsigned short KTs[2][32][KEYS];      // 8 KB
    __shared__ alignas(16) unsigned short Vs[2][KEYS][C_];       // 64 KB
    __shared__ alignas(16) unsigned short Ps[AT_WAVES][16][KEYS];// 8 KB

    const int tid  = threadIdx.x;
    const int lane = tid & 31;
    const int wave = tid >> 5;
    const int half = lane >> 4;
    const int l15  = lane & 15;

    const int b     = blockIdx.x >> 6;
    const int qBase = ((blockIdx.x & 63) * AT_WAVES + wave) * 16;
    const float g   = gammaPtr[0];

    // Q A-fragment, held in VGPRs for the whole loop
    FragBF qf;
    {
        const unsigned short* qrow = Q + ((size_t)b * N_ + qBase + l15) * CQ_;
        const int kA = half * 8;
        qf.q[0] = *(const uint4*)(qrow + kA);
        qf.q[1] = *(const uint4*)(qrow + kA + 16);
    }

    float Mrow[8], Lrow[8];
    v8f O[16];
#pragma unroll
    for (int r = 0; r < 8; ++r) { Mrow[r] = -3.0e38f; Lrow[r] = 0.f; }
#pragma unroll
    for (int t = 0; t < 16; ++t) O[t] = (v8f){0.f,0.f,0.f,0.f,0.f,0.f,0.f,0.f};

    const unsigned short* KTb = KT + (size_t)b * CQ_ * N_;
    const unsigned short* Vb  = V  + (size_t)b * N_ * C_;

    const int krow = tid >> 2;            // KT row (dim 0..31)
    const int kch  = (tid & 3) * 16;      // 16 keys = 32B per thread
    const int vkey = tid >> 1;            // V row (key 0..63)
    const int vch  = (tid & 1) * 128;     // 128 channels = 256B per thread

    auto stage = [&](int tile) {
        const int kb = tile * KEYS;
        const int nb = tile & 1;
        stage_tile(KTb + (size_t)krow * N_ + kb + kch, &KTs[nb][krow][kch],
                   Vb + (size_t)(kb + vkey) * C_ + vch, &Vs[nb][vkey][vch]);
    };

    stage(0);                              // prologue: tile 0 in flight

    const v8f zero = (v8f){0.f,0.f,0.f,0.f,0.f,0.f,0.f,0.f};

    for (int i = 0; i < KITERS; ++i) {
        if (i + 1 < KITERS) {
            stage(i + 1);                                  // issue next tile
            asm volatile("s_wait_asynccnt 0x12" ::: "memory"); // tile i done
        } else {
            asm volatile("s_wait_asynccnt 0x0" ::: "memory");  // drain tail
        }
        if (i + 2 < KITERS)                                // warm L2 ahead
            __builtin_prefetch(Vb + (size_t)((i + 2) * KEYS + vkey) * C_ + vch, 0, 1);
        __syncthreads();
        const int buf = i & 1;

        // ---- S = Q . K^T : four 16-key fragments ----
        v8f S[4];
#pragma unroll
        for (int j = 0; j < 4; ++j) {
            FragBF kf;
            for (int s = 0; s < 16; ++s)
                kf.v[s] = __builtin_bit_cast(bf16_t, KTs[buf][half * 8 + (s & 7) + ((s >> 3) << 4)][j * 16 + l15]);
            S[j] = __builtin_amdgcn_wmma_f32_16x16x32_bf16(
                       false, qf.v, false, kf.v, (short)0, zero, false, false);
        }

        // ---- online softmax over 64 keys ----
        float scale[8];
#pragma unroll
        for (int r = 0; r < 8; ++r) {
            float rm = fmaxf(fmaxf(S[0][r], S[1][r]), fmaxf(S[2][r], S[3][r]));
#pragma unroll
            for (int off = 8; off >= 1; off >>= 1)
                rm = fmaxf(rm, __shfl_xor(rm, off, 32));
            const float nm = fmaxf(Mrow[r], rm);
            scale[r] = __expf(Mrow[r] - nm);
            Mrow[r]  = nm;
            float rs = 0.f;
#pragma unroll
            for (int j = 0; j < 4; ++j) {
                const float p = __expf(S[j][r] - nm);
                rs += p;
                Ps[wave][r + 8 * half][l15 + 16 * j] = bf1(p);
            }
#pragma unroll
            for (int off = 8; off >= 1; off >>= 1)
                rs += __shfl_xor(rs, off, 32);
            Lrow[r] = Lrow[r] * scale[r] + rs;
        }
#pragma unroll
        for (int t = 0; t < 16; ++t)
#pragma unroll
            for (int r = 0; r < 8; ++r) O[t][r] *= scale[r];

        // ---- reload P as two A fragments (keys 0..31, 32..63) ----
        FragBF pf0, pf1;
        {
            const int kA = half * 8;
            pf0.q[0] = *(const uint4*)&Ps[wave][l15][kA];
            pf0.q[1] = *(const uint4*)&Ps[wave][l15][kA + 16];
            pf1.q[0] = *(const uint4*)&Ps[wave][l15][32 + kA];
            pf1.q[1] = *(const uint4*)&Ps[wave][l15][48 + kA];
        }

        // ---- O += P . V ----
#pragma unroll
        for (int t = 0; t < 16; ++t) {
            FragBF vf0, vf1;
            for (int s = 0; s < 16; ++s)
                vf0.v[s] = __builtin_bit_cast(bf16_t, Vs[buf][half * 8 + (s & 7) + ((s >> 3) << 4)][t * 16 + l15]);
            O[t] = __builtin_amdgcn_wmma_f32_16x16x32_bf16(
                       false, pf0.v, false, vf0.v, (short)0, O[t], false, false);
            for (int s = 0; s < 16; ++s)
                vf1.v[s] = __builtin_bit_cast(bf16_t, Vs[buf][32 + half * 8 + (s & 7) + ((s >> 3) << 4)][t * 16 + l15]);
            O[t] = __builtin_amdgcn_wmma_f32_16x16x32_bf16(
                       false, pf1.v, false, vf1.v, (short)0, O[t], false, false);
        }
        __syncthreads();   // reads done before this buffer is re-staged
    }

    // ---- epilogue: out = gamma * O / L + residual ----
    float inv[8];
#pragma unroll
    for (int r = 0; r < 8; ++r) inv[r] = 1.0f / Lrow[r];
#pragma unroll
    for (int t = 0; t < 16; ++t) {
#pragma unroll
        for (int r = 0; r < 8; ++r) {
            const int tok = qBase + r + 8 * half;
            const int c   = t * 16 + l15;
            const size_t idx = ((size_t)b * N_ + tok) * C_ + c;
            out[idx] = g * O[t][r] * inv[r] + inputs[idx];
        }
    }
}

// ---------------------------------------------------------------------------
extern "C" void kernel_launch(void* const* d_in, const int* in_sizes, int n_in,
                              void* d_out, int out_size, void* d_ws, size_t ws_size,
                              hipStream_t stream)
{
    const float* x  = (const float*)d_in[0];
    const float* wq = (const float*)d_in[1];
    const float* bq = (const float*)d_in[2];
    const float* wk = (const float*)d_in[3];
    const float* bk = (const float*)d_in[4];
    const float* wv = (const float*)d_in[5];
    const float* bv = (const float*)d_in[6];
    const float* gm = (const float*)d_in[7];
    float* out = (float*)d_out;

    const int M = B_ * N_;                 // 16384

    // workspace (bf16): Xbf 8MB | Wq 16KB | Wk 16KB | Wv 128KB | Q 1MB | KT 1MB | V 8MB
    char* ws = (char*)d_ws;
    size_t off = 0;
    unsigned short* Xbf = (unsigned short*)(ws + off); off += (size_t)M * C_ * 2;
    unsigned short* Wq  = (unsigned short*)(ws + off); off += (size_t)C_ * CQ_ * 2;
    unsigned short* Wk  = (unsigned short*)(ws + off); off += (size_t)C_ * CQ_ * 2;
    unsigned short* Wv  = (unsigned short*)(ws + off); off += (size_t)C_ * C_ * 2;
    unsigned short* Q   = (unsigned short*)(ws + off); off += (size_t)M * CQ_ * 2;
    unsigned short* KT  = (unsigned short*)(ws + off); off += (size_t)M * CQ_ * 2;
    unsigned short* V   = (unsigned short*)(ws + off);

    // 1) one-pass conversions to bf16
    {
        int n8 = M * C_ / 8;
        cvt_bf16_kernel<<<dim3((n8 + 255) / 256), dim3(256), 0, stream>>>(x, Xbf, n8);
        n8 = C_ * CQ_ / 8;
        cvt_bf16_kernel<<<dim3((n8 + 255) / 256), dim3(256), 0, stream>>>(wq, Wq, n8);
        cvt_bf16_kernel<<<dim3((n8 + 255) / 256), dim3(256), 0, stream>>>(wk, Wk, n8);
        n8 = C_ * C_ / 8;
        cvt_bf16_kernel<<<dim3((n8 + 255) / 256), dim3(256), 0, stream>>>(wv, Wv, n8);
    }

    // 2) projections (pure WMMA loops)
    dim3 pblk(256);
    qkv_proj_kernel<<<dim3(M / 16 / 8, CQ_ / 16), pblk, 0, stream>>>(Xbf, Wq, bq, Q,  CQ_, 0);
    qkv_proj_kernel<<<dim3(M / 16 / 8, CQ_ / 16), pblk, 0, stream>>>(Xbf, Wk, bk, KT, CQ_, 1);
    qkv_proj_kernel<<<dim3(M / 16 / 8, C_  / 16), pblk, 0, stream>>>(Xbf, Wv, bv, V,  C_,  0);

    // 3) flash attention
    attn_kernel<<<dim3(B_ * (N_ / (16 * AT_WAVES))), dim3(128), 0, stream>>>(
        Q, KT, V, x, gm, out);
}
